// FactorizedAttention_25786983645931
// MI455X (gfx1250) — matrix operand order repaired and checked by reading, not compiled, not run on hardware
//
#include <hip/hip_runtime.h>
#include <hip/hip_bf16.h>
#include <math.h>
#include <stdint.h>

// ---- problem constants ----
#define Bn 4
#define Nn 8192
#define Cn 768
#define Hn 12
#define Dn 64
#define QKVW 2304

typedef __bf16 bf16_t;
typedef __attribute__((ext_vector_type(4)))  __bf16 v4bf;
typedef __attribute__((ext_vector_type(8)))  __bf16 v8bf;
typedef __attribute__((ext_vector_type(16))) __bf16 v16bf;
typedef __attribute__((ext_vector_type(8)))  float  v8f;

__device__ __forceinline__ v16bf cat16(v8bf lo, v8bf hi) {
    return __builtin_shufflevector(lo, hi, 0,1,2,3,4,5,6,7,8,9,10,11,12,13,14,15);
}

__device__ __forceinline__ v8f wmma_bf16(v16bf a, v16bf b, v8f c) {
#if defined(__gfx1250__)
    return __builtin_amdgcn_wmma_f32_16x16x32_bf16(false, a, false, b, (short)0, c, false, false);
#else
    return c;
#endif
}

// ---- CDNA5 async global->LDS staging (ASYNCcnt path), with safe fallback ----
#if defined(__HIP_DEVICE_COMPILE__) && __has_builtin(__builtin_amdgcn_global_load_async_to_lds_b128)
#define HAVE_ASYNC_COPY 1
#else
#define HAVE_ASYNC_COPY 0
#endif

#if HAVE_ASYNC_COPY
// exact pointee type from the builtin's signature:
// '__attribute__((__vector_size__(4 * sizeof(int)))) int *'
typedef int v4i_vs __attribute__((__vector_size__(4 * sizeof(int))));
typedef __attribute__((address_space(1))) v4i_vs* g_v4i_ptr;   // global src
typedef __attribute__((address_space(3))) v4i_vs* l_v4i_ptr;   // LDS dst
#endif

__device__ __forceinline__ void stage16(const bf16_t* g, bf16_t* s) {
#if HAVE_ASYNC_COPY
    __builtin_amdgcn_global_load_async_to_lds_b128(
        (g_v4i_ptr)(uintptr_t)g,
        (l_v4i_ptr)(uint32_t)(uintptr_t)s,
        0, 0);
#else
    *(v8bf*)s = *(const v8bf*)g;
#endif
}

template <int N_>
__device__ __forceinline__ void wait_async() {
#if HAVE_ASYNC_COPY
#if __has_builtin(__builtin_amdgcn_s_wait_asynccnt)
    __builtin_amdgcn_s_wait_asynccnt(N_);
#else
    asm volatile("s_wait_asynccnt %0" :: "i"(N_) : "memory");
#endif
#endif
}

// ---- block reductions (256 threads) ----
__device__ __forceinline__ float block_sum256(float v, float* red) {
    int t = threadIdx.x;
    red[t] = v; __syncthreads();
    #pragma unroll
    for (int s = 128; s > 0; s >>= 1) {
        if (t < s) red[t] += red[t + s];
        __syncthreads();
    }
    float r = red[0]; __syncthreads();
    return r;
}
__device__ __forceinline__ float block_max256(float v, float* red) {
    int t = threadIdx.x;
    red[t] = v; __syncthreads();
    #pragma unroll
    for (int s = 128; s > 0; s >>= 1) {
        if (t < s) red[t] = fmaxf(red[t], red[t + s]);
        __syncthreads();
    }
    float r = red[0]; __syncthreads();
    return r;
}

// ---- f32 -> bf16 weight conversion ----
__global__ void k_cvt_bf16(const float* __restrict__ src, bf16_t* __restrict__ dst, int n) {
    int i = blockIdx.x * blockDim.x + threadIdx.x;
    int stride = gridDim.x * blockDim.x;
    for (; i < n; i += stride) dst[i] = (bf16_t)src[i];
}

// ---- LayerNorm over C=768, one row per 256-thread block, bf16 output ----
__global__ void k_layernorm(const float* __restrict__ x, const float* __restrict__ w,
                            const float* __restrict__ bvec, bf16_t* __restrict__ xnb) {
    __shared__ float red[256];
    const int row = blockIdx.x;
    const int t = threadIdx.x;
    const float* xr = x + (size_t)row * Cn;
    float v0 = xr[t], v1 = xr[t + 256], v2 = xr[t + 512];
    float mu = block_sum256(v0 + v1 + v2, red) * (1.0f / Cn);
    float d0 = v0 - mu, d1 = v1 - mu, d2 = v2 - mu;
    float var = block_sum256(d0 * d0 + d1 * d1 + d2 * d2, red) * (1.0f / Cn);
    float rinv = rsqrtf(var + 1e-5f);
    bf16_t* o = xnb + (size_t)row * Cn;
    o[t]       = (bf16_t)(d0 * rinv * w[t]       + bvec[t]);
    o[t + 256] = (bf16_t)(d1 * rinv * w[t + 256] + bvec[t + 256]);
    o[t + 512] = (bf16_t)(d2 * rinv * w[t + 512] + bvec[t + 512]);
}

// ---- Tiled WMMA GEMM: out(M x Nw) = A(M x 768) * W(Nw x 768)^T + bias ----
// 256 threads (8 waves), block tile 128x128, K staged 32 at a time.
// Double-buffered LDS; tiles staged with GLOBAL_LOAD_ASYNC_TO_LDS_B128
// (ASYNCcnt), overlapping HBM latency with WMMA issue.
// LDS row stride 40 bf16 (20 dwords) -> conflict-free b128 fragment reads.
template <bool BF16OUT>
__global__ void k_gemm(const bf16_t* __restrict__ A, const bf16_t* __restrict__ W,
                       const float* __restrict__ bias, void* __restrict__ outv, int Nw) {
    __shared__ bf16_t sA[2][128 * 40];
    __shared__ bf16_t sB[2][128 * 40];
    const int t = threadIdx.x;
    const int m0 = blockIdx.y * 128;
    const int n0 = blockIdx.x * 128;
    const int lane = t & 31, w = t >> 5;
    const int wm = w >> 2, wn = w & 3;          // 2 x 4 wave grid, each wave 64x32
    const int half = lane >> 4, l16 = lane & 15;
    const int lrow = t >> 1, lhalf = t & 1;     // cooperative staging mapping
    const bf16_t* ga = A + (size_t)(m0 + lrow) * 768 + lhalf * 16;
    const bf16_t* gb = W + (size_t)(n0 + lrow) * 768 + lhalf * 16;
    const int soff = lrow * 40 + lhalf * 16;

    v8f acc[4][2] = {};
    // prologue: stage k-chunk 0 into buffer 0
    stage16(ga,     &sA[0][soff]);
    stage16(ga + 8, &sA[0][soff + 8]);
    stage16(gb,     &sB[0][soff]);
    stage16(gb + 8, &sB[0][soff + 8]);

    for (int i = 0; i < 24; ++i) {
        const int cur = i & 1;
        if (i + 1 < 24) {   // stage next chunk into the other buffer
            const int k0 = (i + 1) * 32;
            stage16(ga + k0,     &sA[cur ^ 1][soff]);
            stage16(ga + k0 + 8, &sA[cur ^ 1][soff + 8]);
            stage16(gb + k0,     &sB[cur ^ 1][soff]);
            stage16(gb + k0 + 8, &sB[cur ^ 1][soff + 8]);
            wait_async<4>();   // current buffer's 4 copies complete; 4 in flight
        } else {
            wait_async<0>();
        }
        __syncthreads();
        v16bf bfrag[2];
        #pragma unroll
        for (int nj = 0; nj < 2; ++nj) {
            int nr = wn * 32 + nj * 16 + l16;
            v8bf lo = *(const v8bf*)&sB[cur][nr * 40 + half * 8];
            v8bf hi = *(const v8bf*)&sB[cur][nr * 40 + 16 + half * 8];
            bfrag[nj] = cat16(lo, hi);
        }
        #pragma unroll
        for (int mi = 0; mi < 4; ++mi) {
            int mr = wm * 64 + mi * 16 + l16;
            v8bf lo = *(const v8bf*)&sA[cur][mr * 40 + half * 8];
            v8bf hi = *(const v8bf*)&sA[cur][mr * 40 + 16 + half * 8];
            v16bf afrag = cat16(lo, hi);
            acc[mi][0] = wmma_bf16(afrag, bfrag[0], acc[mi][0]);
            acc[mi][1] = wmma_bf16(afrag, bfrag[1], acc[mi][1]);
        }
        __syncthreads();
    }
    #pragma unroll
    for (int nj = 0; nj < 2; ++nj) {
        int ng = n0 + wn * 32 + nj * 16 + l16;
        float bv = bias[ng];
        #pragma unroll
        for (int mi = 0; mi < 4; ++mi) {
            int mg = m0 + wm * 64 + mi * 16 + half * 8;
            #pragma unroll
            for (int i = 0; i < 8; ++i) {
                float v = acc[mi][nj][i] + bv;
                if (BF16OUT) ((bf16_t*)outv)[(size_t)(mg + i) * Nw + ng] = (bf16_t)v;
                else         ((float*)outv)[(size_t)(mg + i) * Nw + ng] = v;
            }
        }
    }
}

// ---- Softmax over sequence axis N for K, in-place on qkv bf16 buffer ----
// One block per (b, h*64+d) column; K region is columns [768, 1536).
// The 50 MB bf16 K region fits in the 192 MB L2, so the strided column
// accesses are L2-resident after first touch.
__global__ void k_softmax_n(bf16_t* __restrict__ qkvb) {
    __shared__ float red[256];
    const int bid = blockIdx.x;
    const int b = bid / Cn, col = bid % Cn;
    const int t = threadIdx.x;
    bf16_t* base = qkvb + (size_t)b * Nn * QKVW + Cn + col;
    float vals[32];
    float m = -1e30f;
    #pragma unroll
    for (int i = 0; i < 32; ++i) {
        vals[i] = (float)base[(size_t)(t + 256 * i) * QKVW];
        m = fmaxf(m, vals[i]);
    }
    m = block_max256(m, red);
    float s = 0.f;
    #pragma unroll
    for (int i = 0; i < 32; ++i) { vals[i] = __expf(vals[i] - m); s += vals[i]; }
    s = block_sum256(s, red);
    float inv = 1.0f / s;
    #pragma unroll
    for (int i = 0; i < 32; ++i)
        base[(size_t)(t + 256 * i) * QKVW] = (bf16_t)(vals[i] * inv);
}

// ---- ctx[b,h] = K_soft^T (64xN) * V (Nx64), scale folded in, bf16 out ----
// 48 blocks of 512 threads (16 waves, one 16x16 tile each). N staged 128 rows
// at a time (4 WMMAs per barrier pair). K/V tiles stored transposed into LDS;
// row stride 136 bf16 (68 dwords, gcd(68,64)=4) -> conflict-free b128 reads.
__global__ void k_ctx(const bf16_t* __restrict__ qkvb, bf16_t* __restrict__ ctxb) {
    __shared__ bf16_t sK[64 * 136];
    __shared__ bf16_t sV[64 * 136];
    const int bh = blockIdx.x;
    const int b = bh / Hn, h = bh % Hn;
    const int t = threadIdx.x;
    const int lane = t & 31, w = t >> 5;
    const int dt = w >> 2, et = w & 3;
    const int half = lane >> 4, l16 = lane & 15;
    const int lrow = t >> 4, lseg = t & 15;   // 32 rows x 16 segs x 4 elems
    const bf16_t* gk = qkvb + ((size_t)b * Nn + lrow) * QKVW + Cn + h * 64 + lseg * 4;

    v8f acc = {};
    for (int c = 0; c < Nn / 128; ++c) {
        v4bf kv[4], vv[4];
        #pragma unroll
        for (int rr = 0; rr < 4; ++rr) {
            const bf16_t* p = gk + (size_t)(c * 128 + rr * 32) * QKVW;
            kv[rr] = *(const v4bf*)p;
            vv[rr] = *(const v4bf*)(p + Cn);   // V region is +768 columns from K
        }
        __syncthreads();
        #pragma unroll
        for (int rr = 0; rr < 4; ++rr) {
            #pragma unroll
            for (int u = 0; u < 4; ++u) {
                sK[(lseg * 4 + u) * 136 + rr * 32 + lrow] = kv[rr][u];  // [d][n]
                sV[(lseg * 4 + u) * 136 + rr * 32 + lrow] = vv[rr][u];  // [e][n]
            }
        }
        __syncthreads();
        const int d = dt * 16 + l16;
        const int e = et * 16 + l16;
        #pragma unroll
        for (int sub = 0; sub < 4; ++sub) {
            v8bf alo = *(const v8bf*)&sK[d * 136 + sub * 32 + half * 8];
            v8bf ahi = *(const v8bf*)&sK[d * 136 + sub * 32 + 16 + half * 8];
            v8bf blo = *(const v8bf*)&sV[e * 136 + sub * 32 + half * 8];
            v8bf bhi = *(const v8bf*)&sV[e * 136 + sub * 32 + 16 + half * 8];
            acc = wmma_bf16(cat16(alo, ahi), cat16(blo, bhi), acc);
        }
    }
    const float scale = 0.125f;   // D^-0.5 folded into ctx
    #pragma unroll
    for (int i = 0; i < 8; ++i) {
        int d = dt * 16 + half * 8 + i;
        int e = et * 16 + l16;
        ctxb[((size_t)bh * 64 + d) * 64 + e] = (bf16_t)(acc[i] * scale);
    }
}

// ---- attn[b,:,h] = Q (Nx64) * ctx (64x64), bf16 out ----
// grid (N/128, B*H); 8 waves of 16 rows each; ctx staged transposed in LDS.
__global__ void k_qctx(const bf16_t* __restrict__ qkvb, const bf16_t* __restrict__ ctxb,
                       bf16_t* __restrict__ attnb) {
    __shared__ bf16_t sC[64 * 72];
    const int nc = blockIdx.x, bh = blockIdx.y;
    const int b = bh / Hn, h = bh % Hn;
    const int t = threadIdx.x;
    #pragma unroll
    for (int r = 0; r < 16; ++r) {
        int lin = t + r * 256;
        int d = lin >> 6, e = lin & 63;
        sC[e * 72 + d] = ctxb[((size_t)bh << 12) + lin];   // transposed: [e][d]
    }
    __syncthreads();
    const int lane = t & 31, w = t >> 5;
    const int half = lane >> 4, l16 = lane & 15;
    const int rowg = nc * 128 + w * 16 + l16;
    const bf16_t* ap = qkvb + ((size_t)b * Nn + rowg) * QKVW + h * 64;  // Q region
    v8f acc[4] = {};
    #pragma unroll
    for (int ks = 0; ks < 2; ++ks) {
        v8bf alo = *(const v8bf*)(ap + ks * 32 + half * 8);
        v8bf ahi = *(const v8bf*)(ap + ks * 32 + 16 + half * 8);
        v16bf afrag = cat16(alo, ahi);
        #pragma unroll
        for (int et = 0; et < 4; ++et) {
            int e = et * 16 + l16;
            v8bf blo = *(const v8bf*)&sC[e * 72 + ks * 32 + half * 8];
            v8bf bhi = *(const v8bf*)&sC[e * 72 + ks * 32 + 16 + half * 8];
            acc[et] = wmma_bf16(afrag, cat16(blo, bhi), acc[et]);
        }
    }
    #pragma unroll
    for (int et = 0; et < 4; ++et) {
        int col = h * 64 + et * 16 + l16;
        #pragma unroll
        for (int i = 0; i < 8; ++i) {
            int m = nc * 128 + w * 16 + half * 8 + i;
            attnb[((size_t)b * Nn + m) * Cn + col] = (bf16_t)acc[et][i];
        }
    }
}

extern "C" void kernel_launch(void* const* d_in, const int* in_sizes, int n_in,
                              void* d_out, int out_size, void* d_ws, size_t ws_size,
                              hipStream_t stream) {
    const float* x      = (const float*)d_in[0];
    const float* ln_w   = (const float*)d_in[1];
    const float* ln_b   = (const float*)d_in[2];
    const float* qkv_w  = (const float*)d_in[3];
    const float* qkv_b  = (const float*)d_in[4];
    const float* proj_w = (const float*)d_in[5];
    const float* proj_b = (const float*)d_in[6];
    float* out = (float*)d_out;

    char* ws = (char*)d_ws;
    size_t off = 0;
    auto wsalloc = [&](size_t bytes) -> void* {
        void* p = ws + off;
        off = (off + bytes + 255) & ~(size_t)255;
        return p;
    };
    bf16_t* xnb   = (bf16_t*)wsalloc((size_t)Bn * Nn * Cn * 2);      // LN output
    bf16_t* qkvb  = (bf16_t*)wsalloc((size_t)Bn * Nn * QKVW * 2);    // q|k|v (k softmaxed in place)
    bf16_t* attnb = (bf16_t*)wsalloc((size_t)Bn * Nn * Cn * 2);      // q @ ctx
    bf16_t* ctxb  = (bf16_t*)wsalloc((size_t)Bn * Hn * Dn * Dn * 2); // scaled ctx
    bf16_t* wqb   = (bf16_t*)wsalloc((size_t)QKVW * Cn * 2);
    bf16_t* wpb   = (bf16_t*)wsalloc((size_t)Cn * Cn * 2);

    k_cvt_bf16<<<512, 256, 0, stream>>>(qkv_w, wqb, QKVW * Cn);
    k_cvt_bf16<<<256, 256, 0, stream>>>(proj_w, wpb, Cn * Cn);
    k_layernorm<<<Bn * Nn, 256, 0, stream>>>(x, ln_w, ln_b, xnb);
    k_gemm<true><<<dim3(QKVW / 128, Bn * Nn / 128), 256, 0, stream>>>(xnb, wqb, qkv_b, qkvb, QKVW);
    k_softmax_n<<<Bn * Cn, 256, 0, stream>>>(qkvb);
    k_ctx<<<Bn * Hn, 512, 0, stream>>>(qkvb, ctxb);
    k_qctx<<<dim3(Nn / 128, Bn * Hn), 256, 0, stream>>>(qkvb, ctxb, attnb);
    k_gemm<false><<<dim3(Cn / 128, Bn * Nn / 128), 256, 0, stream>>>(attnb, wpb, proj_b, out, Cn);
}
